// CSConv_6674379178242
// MI455X (gfx1250) — matrix operand order repaired and checked by reading, not compiled, not run on hardware
//
#include <hip/hip_runtime.h>

#define B_    32
#define CIN   128
#define NPIX  4096      // 64*64
#define COUT  256
#define KDIM  1152      // CIN*9
#define TTEMP 34.0f
#define BNEPS 1e-5f

typedef __attribute__((ext_vector_type(16))) __bf16        v16bf;
typedef __attribute__((ext_vector_type(8)))  float         v8f;
typedef __attribute__((ext_vector_type(4)))  unsigned int  u32x4;

__device__ __forceinline__ unsigned short f2bf(float f) {
    unsigned int u = __float_as_uint(f);
    u += 0x7FFFu + ((u >> 16) & 1u);        // round-to-nearest-even
    return (unsigned short)(u >> 16);
}

// ---------------------------------------------------------------- K1: pool
__global__ void pool_kernel(const float* __restrict__ x, float* __restrict__ pooled) {
    __shared__ float red[256];
    const int bc = blockIdx.x;                 // b*128 + c
    const int t  = threadIdx.x;
    const float* src = x + (size_t)bc * NPIX;
    float s = 0.f;
    for (int p = t; p < NPIX; p += 256) s += src[p];
    red[t] = s; __syncthreads();
    for (int off = 128; off > 0; off >>= 1) {
        if (t < off) red[t] += red[t + off];
        __syncthreads();
    }
    if (t == 0) pooled[bc] = red[0] * (1.f / NPIX);
}

// ---------------------------------------------------------------- K2: routing MLP + softmax + bias
__global__ void route_kernel(const float* __restrict__ pooled, const float* __restrict__ w_fc1,
                             const float* __restrict__ g1, const float* __restrict__ be1,
                             const float* __restrict__ m1, const float* __restrict__ v1,
                             const float* __restrict__ w_fc2, const float* __restrict__ bf2,
                             const float* __restrict__ g2, const float* __restrict__ be2,
                             const float* __restrict__ m2, const float* __restrict__ v2,
                             const float* __restrict__ anchored_b,
                             float* __restrict__ theta, float* __restrict__ bias) {
    __shared__ float sp[128], sh[128], st[512], red[128], smk[4];
    const int b = blockIdx.x, t = threadIdx.x;
    sp[t] = pooled[b * 128 + t];
    __syncthreads();

    // fc1 + bn1 + relu
    float acc = 0.f;
    const float* wr = w_fc1 + t * 128;
    #pragma unroll 4
    for (int c = 0; c < 128; ++c) acc += sp[c] * wr[c];
    acc = (acc - m1[t]) * g1[t] * rsqrtf(v1[t] + BNEPS) + be1[t];
    sh[t] = fmaxf(acc, 0.f);
    __syncthreads();

    // fc2 + bn2, /T
    float loc[4];
    for (int q = 0; q < 4; ++q) {
        const int l = q * 128 + t;
        float a2 = bf2[l];
        const float* wr2 = w_fc2 + (size_t)l * 128;
        #pragma unroll 4
        for (int c = 0; c < 128; ++c) a2 += sh[c] * wr2[c];
        a2 = (a2 - m2[l]) * g2[l] * rsqrtf(v2[l] + BNEPS) + be2[l];
        loc[q] = a2 * (1.f / TTEMP);
    }

    // softmax over all 512
    float mx = fmaxf(fmaxf(loc[0], loc[1]), fmaxf(loc[2], loc[3]));
    red[t] = mx; __syncthreads();
    for (int off = 64; off > 0; off >>= 1) { if (t < off) red[t] = fmaxf(red[t], red[t + off]); __syncthreads(); }
    mx = red[0]; __syncthreads();
    float ssum = 0.f;
    for (int q = 0; q < 4; ++q) { float e = __expf(loc[q] - mx); st[q * 128 + t] = e; ssum += e; }
    red[t] = ssum; __syncthreads();
    for (int off = 64; off > 0; off >>= 1) { if (t < off) red[t] += red[t + off]; __syncthreads(); }
    const float inv = 1.f / red[0]; __syncthreads();
    for (int q = 0; q < 4; ++q) {
        const int l = q * 128 + t;
        const float th = st[l] * inv;
        st[l] = th;
        theta[b * 512 + l] = th;
    }
    __syncthreads();

    // mk[k] = mean_i theta[b,k,i]; bias[b,o] = sum_k mk[k]*anchored_b[k,o]
    for (int k = 0; k < 4; ++k) {
        red[t] = st[k * 128 + t]; __syncthreads();
        for (int off = 64; off > 0; off >>= 1) { if (t < off) red[t] += red[t + off]; __syncthreads(); }
        if (t == 0) smk[k] = red[0] * (1.f / 128.f);
        __syncthreads();
    }
    for (int q = 0; q < 2; ++q) {
        const int o = q * 128 + t;
        float bb = 0.f;
        for (int k = 0; k < 4; ++k) bb += smk[k] * anchored_b[k * 256 + o];
        bias[b * 256 + o] = bb;
    }
}

// ---------------------------------------------------------------- K3: mix weights -> bf16 A matrix
__global__ void mix_kernel(const float* __restrict__ theta, const float* __restrict__ aw,
                           unsigned short* __restrict__ wmix) {
    const int blk = blockIdx.x;                // b*256 + o
    const int b = blk >> 8, o = blk & 255;
    const int t = threadIdx.x;                 // 288 threads, 4 elems each
    const float* th = theta + b * 512;
    #pragma unroll
    for (int q = 0; q < 4; ++q) {
        const int k = q * 288 + t;             // 0..1151
        const int i = k / 9;
        const int r = k - i * 9;
        float s = 0.f;
        #pragma unroll
        for (int kk = 0; kk < 4; ++kk)
            s += th[kk * 128 + i] * aw[(((size_t)kk * 256 + o) * 128 + i) * 9 + r];
        wmix[(size_t)blk * KDIM + k] = f2bf(s);
    }
}

// ---------------------------------------------------------------- K4: implicit-GEMM conv via bf16 WMMA
__global__ void __launch_bounds__(256)
csconv_gemm(const float* __restrict__ x, const unsigned short* __restrict__ wmix,
            const float* __restrict__ bias, float* __restrict__ out) {
    __shared__ __align__(16) unsigned short sA[2][128][40];  // [m][k], pad 32->40
    __shared__ __align__(16) unsigned short sB[2][128][40];  // [pixel][k]

    const int t     = threadIdx.x;
    const int nt    = blockIdx.x;      // pixel tile 0..31
    const int mt    = blockIdx.y;      // cout  tile 0..1
    const int b     = blockIdx.z;
    const int Pbase = nt * 128;
    const int Mbase = mt * 128;
    const unsigned short* Ag = wmix + ((size_t)b * COUT + Mbase) * KDIM;
    const float* xb = x + (size_t)b * CIN * NPIX;

    auto stage = [&](int buf, int kt) {
        const int k0 = kt * 32;
        // A tile: 128 rows x 32 bf16, contiguous rows -> 2x b128 per thread
        {
            const int row  = t >> 1;
            const int half = (t & 1) * 16;
            const unsigned short* src = Ag + (size_t)row * KDIM + k0 + half;
            u32x4 q0 = *(const u32x4*)(src);
            u32x4 q1 = *(const u32x4*)(src + 8);
            *(u32x4*)(&sA[buf][row][half])     = q0;
            *(u32x4*)(&sA[buf][row][half + 8]) = q1;
        }
        // B tile: im2col gather (fp32 -> bf16), coalesced along pixel dim
        #pragma unroll 4
        for (int it = 0; it < 16; ++it) {
            const int e  = t + it * 256;       // 0..4095
            const int kk = e >> 7;             // 0..31
            const int p  = e & 127;
            const int k  = k0 + kk;
            const int i  = k / 9;
            const int r  = k - i * 9;
            const int dy = r / 3 - 1;
            const int dx = r - (r / 3) * 3 - 1;
            const int pg = Pbase + p;
            const int y  = (pg >> 6) + dy;
            const int xx = (pg & 63) + dx;
            float v = 0.f;
            if ((unsigned)y < 64u && (unsigned)xx < 64u)
                v = xb[(size_t)i * NPIX + y * 64 + xx];
            sB[buf][p][kk] = f2bf(v);
        }
    };

    const int lane = t & 31;
    const int wv   = t >> 5;           // 8 waves
    const int wm   = wv >> 2;          // 0..1 -> 64 rows
    const int wn   = wv & 3;           // 0..3 -> 32 cols
    const int rlo  = lane & 15;
    const int hiA  = lane >> 4;        // 0/1 half-wave
    const int kbA  = hiA * 8;          // unused: A is also read with kbB (K 0-15/16-31)
    const int kbB  = hiA * 16;         // B: K 0-15/16-31

    v8f acc[4][2] = {};                // 64x32 per wave = 8 WMMA tiles

    union Frag { v16bf v; u32x4 q[2]; };

    stage(0, 0);
    __syncthreads();

    for (int kt = 0; kt < 36; ++kt) {  // K = 36*32 = 1152
        const int cur = kt & 1;
        if (kt + 1 < 36) stage(cur ^ 1, kt + 1);

        Frag fb[2];
        #pragma unroll
        for (int c = 0; c < 2; ++c) {
            const int prow = wn * 32 + c * 16 + rlo;
            fb[c].q[0] = *(const u32x4*)(&sB[cur][prow][kbB]);
            fb[c].q[1] = *(const u32x4*)(&sB[cur][prow][kbB + 8]);
        }
        #pragma unroll
        for (int a = 0; a < 4; ++a) {
            const int mrow = wm * 64 + a * 16 + rlo;
            Frag fa;
            fa.q[0] = *(const u32x4*)(&sA[cur][mrow][kbB]);
            fa.q[1] = *(const u32x4*)(&sA[cur][mrow][kbB + 8]);
            #pragma unroll
            for (int c = 0; c < 2; ++c) {
                acc[a][c] = __builtin_amdgcn_wmma_f32_16x16x32_bf16(
                    false, fa.v, false, fb[c].v, (short)0, acc[a][c], false, false);
            }
        }
        __syncthreads();
    }

    // epilogue: C/D layout — lane%16 = col, lanes>=16 hold rows +8, VGPR g = row
    #pragma unroll
    for (int a = 0; a < 4; ++a) {
        const int obase = Mbase + wm * 64 + a * 16 + hiA * 8;
        #pragma unroll
        for (int c = 0; c < 2; ++c) {
            const int p = Pbase + wn * 32 + c * 16 + rlo;
            #pragma unroll
            for (int g = 0; g < 8; ++g) {
                const int o = obase + g;
                out[((size_t)b * COUT + o) * NPIX + p] = acc[a][c][g] + bias[b * COUT + o];
            }
        }
    }
}

// ---------------------------------------------------------------- launch
extern "C" void kernel_launch(void* const* d_in, const int* in_sizes, int n_in,
                              void* d_out, int out_size, void* d_ws, size_t ws_size,
                              hipStream_t stream) {
    (void)in_sizes; (void)n_in; (void)out_size; (void)ws_size;
    const float* x    = (const float*)d_in[0];
    const float* wfc1 = (const float*)d_in[1];
    const float* g1   = (const float*)d_in[2];
    const float* be1  = (const float*)d_in[3];
    const float* m1   = (const float*)d_in[4];
    const float* v1   = (const float*)d_in[5];
    const float* wfc2 = (const float*)d_in[6];
    const float* bf2  = (const float*)d_in[7];
    const float* g2   = (const float*)d_in[8];
    const float* be2  = (const float*)d_in[9];
    const float* m2   = (const float*)d_in[10];
    const float* v2   = (const float*)d_in[11];
    const float* aw   = (const float*)d_in[12];
    const float* ab   = (const float*)d_in[13];
    float* out = (float*)d_out;

    char* ws = (char*)d_ws;
    float*          theta  = (float*)(ws);                            // 32*512 f32  = 64 KB
    float*          pooled = (float*)(ws + 65536);                    // 4096  f32   = 16 KB
    float*          bias   = (float*)(ws + 65536 + 16384);            // 32*256 f32  = 32 KB
    unsigned short* wmix   = (unsigned short*)(ws + 65536 + 16384 + 32768); // 18.9 MB bf16

    pool_kernel<<<dim3(B_ * CIN), dim3(256), 0, stream>>>(x, pooled);
    route_kernel<<<dim3(B_), dim3(128), 0, stream>>>(pooled, wfc1, g1, be1, m1, v1,
                                                     wfc2, bf2, g2, be2, m2, v2, ab,
                                                     theta, bias);
    mix_kernel<<<dim3(B_ * COUT), dim3(288), 0, stream>>>(theta, aw, wmix);
    csconv_gemm<<<dim3(32, 2, B_), dim3(256), 0, stream>>>(x, wmix, bias, out);
}
